// GramAnchoringObjective_44779329028102
// MI455X (gfx1250) — compile-verified, matrix operand order checked by reading, not run on hardware
//
#include <hip/hip_runtime.h>
#include <hip/hip_bf16.h>

// Problem constants (from reference): N = 16384 rows, D = 768, P = 256, bt = 64
#define KD   768
#define KP   256
#define KBT  64
#define KN   (KBT * KP)      // 16384
#define KMACRO 36            // upper-triangular 32x32 macro-tiles of a 256x256 gram (8*9/2)
#define KEPS 1e-6f

typedef __attribute__((ext_vector_type(16))) _Float16 v16h;
typedef __attribute__((ext_vector_type(8)))  float    v8f;

union Frag {
    v16h v;
    uint4 q[2];
};

__device__ __forceinline__ v16h loadFrag(const _Float16* p) {
    Frag f;
    f.q[0] = *(const uint4*)(p);
    f.q[1] = *(const uint4*)(p + 16);
    return f.v;
}

__device__ __forceinline__ float warpSum(float v) {
    #pragma unroll
    for (int o = 16; o > 0; o >>= 1) v += __shfl_xor(v, o, 32);
    return v;
}

// ---------------------------------------------------------------------------
// Kernel 1: per-row L2 normalize input & target -> f16, and patch_loss[row]
// One 256-thread block per row (768 = 3 * 256 elements).
// ---------------------------------------------------------------------------
__global__ void normalize_kernel(const float* __restrict__ inp,
                                 const float* __restrict__ tgt,
                                 _Float16* __restrict__ Xn,   // normalized input, f16
                                 _Float16* __restrict__ Tn,   // normalized target, f16
                                 float* __restrict__ patch_loss)
{
    const int row = blockIdx.x;
    const int tid = threadIdx.x;            // 0..255
    const float* xi = inp + (size_t)row * KD;
    const float* ti = tgt + (size_t)row * KD;

    float x0 = xi[tid], x1 = xi[tid + 256], x2 = xi[tid + 512];
    float t0 = ti[tid], t1 = ti[tid + 256], t2 = ti[tid + 512];

    float ssx = x0 * x0 + x1 * x1 + x2 * x2;
    float sst = t0 * t0 + t1 * t1 + t2 * t2;
    float d0 = x0 - t0, d1 = x1 - t1, d2 = x2 - t2;
    float sd = d0 * d0 + d1 * d1 + d2 * d2;

    ssx = warpSum(ssx); sst = warpSum(sst); sd = warpSum(sd);

    __shared__ float sm[3][8];
    __shared__ float res[3];
    const int lane = tid & 31, wid = tid >> 5;
    if (lane == 0) { sm[0][wid] = ssx; sm[1][wid] = sst; sm[2][wid] = sd; }
    __syncthreads();
    if (tid == 0) {
        float a = 0.f, b = 0.f, c = 0.f;
        #pragma unroll
        for (int i = 0; i < 8; ++i) { a += sm[0][i]; b += sm[1][i]; c += sm[2][i]; }
        res[0] = a; res[1] = b; res[2] = c;
    }
    __syncthreads();

    const float invx = 1.0f / fmaxf(sqrtf(res[0]), 1e-12f);
    const float invt = 1.0f / fmaxf(sqrtf(res[1]), 1e-12f);

    _Float16* xo = Xn + (size_t)row * KD;
    _Float16* to = Tn + (size_t)row * KD;
    xo[tid]       = (_Float16)(x0 * invx);
    xo[tid + 256] = (_Float16)(x1 * invx);
    xo[tid + 512] = (_Float16)(x2 * invx);
    to[tid]       = (_Float16)(t0 * invt);
    to[tid + 256] = (_Float16)(t1 * invt);
    to[tid + 512] = (_Float16)(t2 * invt);

    if (tid == 0) patch_loss[row] = res[2] * (1.0f / (float)KD);
}

// ---------------------------------------------------------------------------
// Kernel 2: per-batch column sums of tg_n: bs[b][d] = sum_p Tn[b,p,d]
// ---------------------------------------------------------------------------
__global__ void batchsum_kernel(const _Float16* __restrict__ Tn,
                                float* __restrict__ bs)
{
    const int b = blockIdx.x;
    const int d = threadIdx.x;              // 0..767
    const _Float16* p = Tn + (size_t)b * KP * KD + d;
    float s = 0.f;
    for (int q = 0; q < KP; ++q) s += (float)p[(size_t)q * KD];
    bs[b * KD + d] = s;
}

// ---------------------------------------------------------------------------
// Kernel 3: global mean patch: gm[d] = (1/N) * sum_b bs[b][d]
// ---------------------------------------------------------------------------
__global__ void gmean_kernel(const float* __restrict__ bs,
                             float* __restrict__ gm)
{
    const int d = blockIdx.x * blockDim.x + threadIdx.x;
    if (d >= KD) return;
    float s = 0.f;
    for (int b = 0; b < KBT; ++b) s += bs[b * KD + d];
    gm[d] = s * (1.0f / (float)KN);
}

// ---------------------------------------------------------------------------
// Kernel 4: weights + sim_loss.
// tf[b,p]  = 1/( tg_n[b,p] . (bs[b]/P) + eps )     (row-mean of gram_gt, no gram needed)
// idf[b,p] = 1/( tg_n[b,p] . gm        + eps )
// w = tf*idf, normalized per batch; sim partial = sum_p w * patch_loss.
// One 256-thread block per batch.
// ---------------------------------------------------------------------------
__global__ void weights_kernel(const _Float16* __restrict__ Tn,
                               const float* __restrict__ bs,
                               const float* __restrict__ gm,
                               const float* __restrict__ patch_loss,
                               float* __restrict__ w,
                               float* __restrict__ simacc)
{
    const int b = blockIdx.x;
    const int p = threadIdx.x;              // 0..255
    const _Float16* tr = Tn + ((size_t)b * KP + p) * KD;
    const float* bsb = bs + b * KD;

    float dt = 0.f, dg = 0.f;
    for (int d = 0; d < KD; ++d) {
        float v = (float)tr[d];
        dt += v * bsb[d];
        dg += v * gm[d];
    }
    const float tf  = 1.0f / (dt * (1.0f / (float)KP) + KEPS);
    const float idf = 1.0f / (dg + KEPS);
    float wraw = tf * idf;

    // block sum of wraw
    __shared__ float sm[8];
    __shared__ float tot;
    float s = warpSum(wraw);
    const int lane = p & 31, wid = p >> 5;
    if (lane == 0) sm[wid] = s;
    __syncthreads();
    if (p == 0) {
        float a = 0.f;
        #pragma unroll
        for (int i = 0; i < 8; ++i) a += sm[i];
        tot = a;
    }
    __syncthreads();

    const float wn = wraw / (tot + KEPS);
    w[b * KP + p] = wn;

    float sl = wn * patch_loss[b * KP + p];
    sl = warpSum(sl);
    if (lane == 0) sm[wid] = sl;
    __syncthreads();
    if (p == 0) {
        float a = 0.f;
        #pragma unroll
        for (int i = 0; i < 8; ++i) a += sm[i];
        atomicAdd(simacc, a);
    }
}

// ---------------------------------------------------------------------------
// Kernel 5: WMMA gram loss with 2x2 register blocking.
// One wave per (batch, upper-tri 32x32 macro-tile). Per K-step: 8 fragment
// loads (16x b128) feed 8 v_wmma_f32_16x16x32_f16 (4 pred + 4 gt sub-tiles)
// -> 2:1 load:WMMA, 2x the FLOP/byte of the unblocked version.
// Epilogue: sum w_p*w_q*(pred-gt)^2; off-diagonal macro-tiles counted twice.
//
// Fragment gather for X·Xᵀ from row-major f16 X (per ISA 16-bit A/B layouts):
//   lane l: row = tile0 + (l&15);   two 16B chunks per K-step kb:
//     halves [kb + (l>=16?8:0) .. +7]  and  [kb + 16 + (l>=16?8:0) .. +7]
// Identical pattern for A (rows p) and B (rows q, consumed column-major).
// ---------------------------------------------------------------------------
__global__ void __launch_bounds__(256)
gram_loss_kernel(const _Float16* __restrict__ Xn,
                 const _Float16* __restrict__ Tn,
                 const float* __restrict__ w,
                 float* __restrict__ gacc)
{
    const int lane = threadIdx.x & 31;
    const int gw = blockIdx.x * (blockDim.x >> 5) + (threadIdx.x >> 5);

    const int b = gw / KMACRO;
    int t = gw - b * KMACRO;
    int mtp = 0;
    while (t >= 8 - mtp) { t -= 8 - mtp; ++mtp; }   // wave-uniform, <=8 iters
    const int mtq = mtp + t;
    const int p0 = mtp * 32, q0 = mtq * 32;

    const int half = lane >> 4;
    const int lp   = lane & 15;
    const size_t base = (size_t)b * KP * KD;
    const size_t lo   = (size_t)half * 8;

    const _Float16* pXA0 = Xn + base + (size_t)(p0 + lp) * KD + lo;
    const _Float16* pXA1 = pXA0 + (size_t)16 * KD;
    const _Float16* pXB0 = Xn + base + (size_t)(q0 + lp) * KD + lo;
    const _Float16* pXB1 = pXB0 + (size_t)16 * KD;
    const _Float16* pTA0 = Tn + base + (size_t)(p0 + lp) * KD + lo;
    const _Float16* pTA1 = pTA0 + (size_t)16 * KD;
    const _Float16* pTB0 = Tn + base + (size_t)(q0 + lp) * KD + lo;
    const _Float16* pTB1 = pTB0 + (size_t)16 * KD;

    v8f cp00 = {}, cp01 = {}, cp10 = {}, cp11 = {};
    v8f cg00 = {}, cg01 = {}, cg10 = {}, cg11 = {};

    #pragma unroll 2
    for (int kb = 0; kb < KD; kb += 32) {
        v16h xa0 = loadFrag(pXA0 + kb);
        v16h xa1 = loadFrag(pXA1 + kb);
        v16h xb0 = loadFrag(pXB0 + kb);
        v16h xb1 = loadFrag(pXB1 + kb);
        v16h ta0 = loadFrag(pTA0 + kb);
        v16h ta1 = loadFrag(pTA1 + kb);
        v16h tb0 = loadFrag(pTB0 + kb);
        v16h tb1 = loadFrag(pTB1 + kb);

        cp00 = __builtin_amdgcn_wmma_f32_16x16x32_f16(false, xa0, false, xb0, (short)0, cp00, false, false);
        cp01 = __builtin_amdgcn_wmma_f32_16x16x32_f16(false, xa0, false, xb1, (short)0, cp01, false, false);
        cp10 = __builtin_amdgcn_wmma_f32_16x16x32_f16(false, xa1, false, xb0, (short)0, cp10, false, false);
        cp11 = __builtin_amdgcn_wmma_f32_16x16x32_f16(false, xa1, false, xb1, (short)0, cp11, false, false);
        cg00 = __builtin_amdgcn_wmma_f32_16x16x32_f16(false, ta0, false, tb0, (short)0, cg00, false, false);
        cg01 = __builtin_amdgcn_wmma_f32_16x16x32_f16(false, ta0, false, tb1, (short)0, cg01, false, false);
        cg10 = __builtin_amdgcn_wmma_f32_16x16x32_f16(false, ta1, false, tb0, (short)0, cg10, false, false);
        cg11 = __builtin_amdgcn_wmma_f32_16x16x32_f16(false, ta1, false, tb1, (short)0, cg11, false, false);
    }

    // C/D layout: VGPR k holds element (M = k + 8*half, N = lane&15) of each sub-tile
    const float wq0 = w[b * KP + q0 + lp];
    const float wq1 = w[b * KP + q0 + 16 + lp];
    const int rp0 = b * KP + p0 + half * 8;  // rows of sub-tiles i=0
    const int rp1 = rp0 + 16;                // rows of sub-tiles i=1

    float acc = 0.f;
    #pragma unroll
    for (int k = 0; k < 8; ++k) {
        const float wp0 = w[rp0 + k];
        const float wp1 = w[rp1 + k];
        float d;
        d = cp00[k] - cg00[k]; acc += wp0 * wq0 * d * d;
        d = cp01[k] - cg01[k]; acc += wp0 * wq1 * d * d;
        d = cp10[k] - cg10[k]; acc += wp1 * wq0 * d * d;
        d = cp11[k] - cg11[k]; acc += wp1 * wq1 * d * d;
    }
    if (mtp != mtq) acc *= 2.0f;   // symmetric gram: off-diagonal macros counted twice

    acc = warpSum(acc);
    if (lane == 0) atomicAdd(gacc, acc);
}

// ---------------------------------------------------------------------------
// Kernel 6: out = sim_loss + 10 * gram_loss  (both means over bt=64)
// ---------------------------------------------------------------------------
__global__ void finalize_kernel(const float* __restrict__ acc,
                                float* __restrict__ out)
{
    out[0] = acc[0] * (1.0f / (float)KBT) + 10.0f * acc[1] * (1.0f / (float)KBT);
}

// ---------------------------------------------------------------------------
extern "C" void kernel_launch(void* const* d_in, const int* in_sizes, int n_in,
                              void* d_out, int out_size, void* d_ws, size_t ws_size,
                              hipStream_t stream) {
    const float* inp = (const float*)d_in[0];
    const float* tgt = (const float*)d_in[1];
    float* out = (float*)d_out;

    char* ws = (char*)d_ws;
    size_t off = 0;
    _Float16* Tn = (_Float16*)(ws + off); off += (size_t)KN * KD * sizeof(_Float16);
    _Float16* Xn = (_Float16*)(ws + off); off += (size_t)KN * KD * sizeof(_Float16);
    float* bs  = (float*)(ws + off); off += (size_t)KBT * KD * sizeof(float);
    float* gm  = (float*)(ws + off); off += (size_t)KD * sizeof(float);
    float* pl  = (float*)(ws + off); off += (size_t)KN * sizeof(float);
    float* wgt = (float*)(ws + off); off += (size_t)KN * sizeof(float);
    float* acc = (float*)(ws + off); off += 2 * sizeof(float);  // [0]=sim, [1]=gram

    // zero scalar accumulators every call (graph-capture-safe memset node)
    hipMemsetAsync(acc, 0, 2 * sizeof(float), stream);

    normalize_kernel<<<KN, 256, 0, stream>>>(inp, tgt, Xn, Tn, pl);
    batchsum_kernel<<<KBT, KD, 0, stream>>>(Tn, bs);
    gmean_kernel<<<3, 256, 0, stream>>>(bs, gm);
    weights_kernel<<<KBT, 256, 0, stream>>>(Tn, bs, gm, pl, wgt, acc);

    // 64 batches * 36 macro-tiles = 2304 waves; 8 waves (256 threads) per block
    gram_loss_kernel<<<(KBT * KMACRO) / 8, 256, 0, stream>>>(Xn, Tn, wgt, acc + 1);

    finalize_kernel<<<1, 1, 0, stream>>>(acc, out);
}
